// AttnBlock_31267361915574
// MI455X (gfx1250) — matrix operand order repaired and checked by reading, not compiled, not run on hardware
//
#include <hip/hip_runtime.h>

#define B_DIM 4
#define C_DIM 512
#define N_TOK 4096
#define NGRP  32
#define Q_SCALE 0.04419417382415922f  /* 512^-0.5 */

// padded LDS row pitches (halves) to kill bank conflicts:
// 520*2B = 1040B == 16 mod 256  -> 16 distinct bank quads across lanes
// 40*2B  =   80B -> offsets {0,80,160,...} all distinct mod 256
#define QROW 520
#define VROW 40

typedef __attribute__((ext_vector_type(16))) _Float16 v16h;
typedef __attribute__((ext_vector_type(8)))  float    v8f;

union FragU { v16h h; uint4 u[2]; };

__device__ __forceinline__ v16h ld_frag(const _Float16* p0, const _Float16* p1) {
  FragU f;
  f.u[0] = *(const uint4*)p0;
  f.u[1] = *(const uint4*)p1;
  return f.h;
}

__device__ __forceinline__ v8f wmma_f16(v16h a, v16h b, v8f c) {
  return __builtin_amdgcn_wmma_f32_16x16x32_f16(false, a, false, b, (short)0, c,
                                                false, false);
}

// CDNA5 async global->LDS copy (ASYNCcnt-tracked), 16B per lane.
__device__ __forceinline__ void async_b128(unsigned lds_addr,
                                           const _Float16* gaddr) {
  asm volatile("global_load_async_to_lds_b128 %0, %1, off"
               :: "v"(lds_addr), "v"(gaddr)
               : "memory");
}
__device__ __forceinline__ void wait_async0() {
  asm volatile("s_wait_asynccnt 0" ::: "memory");
}

// ---------------------------------------------------------------------------
// K0: fp32 -> f16 weight conversion
// ---------------------------------------------------------------------------
__global__ __launch_bounds__(256) void wconv_kernel(const float* __restrict__ src,
                                                    _Float16* __restrict__ dst,
                                                    int n) {
  int i = blockIdx.x * 256 + threadIdx.x;
  if (i < n) dst[i] = (_Float16)src[i];
}

// ---------------------------------------------------------------------------
// K1: GroupNorm statistics. One block per (b, group) = 128 blocks.
// ---------------------------------------------------------------------------
__global__ __launch_bounds__(256) void gn_stats_kernel(const float* __restrict__ x,
                                                       float* __restrict__ stats) {
  __shared__ float red_s[8];
  __shared__ float red_q[8];
  const int tid = threadIdx.x;
  const int lane = tid & 31;
  const int wid = tid >> 5;
  const size_t base = (size_t)blockIdx.x * 65536;  // 16 ch * 4096 tok
  const float4* src = (const float4*)(x + base);
  float s = 0.f, q = 0.f;
  for (int i = tid; i < 16384; i += 256) {
    float4 v = src[i];
    s += v.x + v.y + v.z + v.w;
    q += v.x * v.x + v.y * v.y + v.z * v.z + v.w * v.w;
  }
  #pragma unroll
  for (int m = 16; m >= 1; m >>= 1) {
    s += __shfl_xor(s, m, 32);
    q += __shfl_xor(q, m, 32);
  }
  if (lane == 0) { red_s[wid] = s; red_q[wid] = q; }
  __syncthreads();
  if (tid == 0) {
    float ts = 0.f, tq = 0.f;
    #pragma unroll
    for (int i = 0; i < 8; ++i) { ts += red_s[i]; tq += red_q[i]; }
    float mean = ts * (1.0f / 65536.0f);
    float var = tq * (1.0f / 65536.0f) - mean * mean;
    stats[blockIdx.x * 2 + 0] = mean;
    stats[blockIdx.x * 2 + 1] = rsqrtf(var + 1e-6f);
  }
}

// ---------------------------------------------------------------------------
// K2: normalize + transpose: hn[b][n][c] f16 = GN(x[b][c][n]).
// ---------------------------------------------------------------------------
__global__ __launch_bounds__(256) void hn_kernel(const float* __restrict__ x,
                                                 const float* __restrict__ stats,
                                                 const float* __restrict__ gnw,
                                                 const float* __restrict__ gnb,
                                                 _Float16* __restrict__ hn) {
  __shared__ float tile[32][33];
  const int bx = blockIdx.x;
  const int b = bx >> 11;
  const int rem = bx & 2047;
  const int c0 = (rem >> 7) << 5;
  const int n0 = (rem & 127) << 5;
  const int tx = threadIdx.x;
  const int ty = threadIdx.y;
  #pragma unroll
  for (int k = 0; k < 4; ++k) {
    int cl = ty + 8 * k;
    int c = c0 + cl;
    float mean = stats[(b * NGRP + (c >> 4)) * 2 + 0];
    float rstd = stats[(b * NGRP + (c >> 4)) * 2 + 1];
    float v = x[((size_t)b * C_DIM + c) * N_TOK + n0 + tx];
    tile[cl][tx] = (v - mean) * rstd * gnw[c] + gnb[c];
  }
  __syncthreads();
  #pragma unroll
  for (int k = 0; k < 4; ++k) {
    int nl = ty + 8 * k;
    hn[((size_t)b * N_TOK + n0 + nl) * C_DIM + c0 + tx] = (_Float16)tile[tx][nl];
  }
}

// ---------------------------------------------------------------------------
// K3: projection GEMM via WMMA. Block = 64 tokens (4 tiles share A-fragments),
// 4 waves each own 8 c_out tiles. transpose_store: 0 -> [b][n][c], 1 -> [b][c][n]
// ---------------------------------------------------------------------------
__global__ __launch_bounds__(128) void proj_kernel(const _Float16* __restrict__ hn,
                                                   const _Float16* __restrict__ w16,
                                                   const float* __restrict__ bias,
                                                   _Float16* __restrict__ outp,
                                                   float scale, int transpose_store) {
  extern __shared__ __align__(16) char smem[];
  _Float16* hs = (_Float16*)smem;  // [64][QROW]
  const int tid = threadIdx.x;
  const int lane = tid & 31;
  const int wid = tid >> 5;
  const int hl = lane >> 4;
  const int ln = lane & 15;
  const int aoff = hl * 8;
  const int koff = hl * 16;
  const int b = blockIdx.x >> 6;
  const int i0 = (blockIdx.x & 63) << 6;  // 64 tokens per block

  for (int t = tid; t < 64 * 64; t += 128) {  // 64 rows x 64 16B-segs
    int r = t >> 6, seg = (t & 63) * 8;
    *(uint4*)(hs + r * QROW + seg) =
        *(const uint4*)(hn + ((size_t)b * N_TOK + i0 + r) * C_DIM + seg);
  }
  __syncthreads();

  for (int ct = wid; ct < 32; ct += 4) {
    const v8f vzero = {0.f, 0.f, 0.f, 0.f, 0.f, 0.f, 0.f, 0.f};
    v8f acc[4];
    #pragma unroll
    for (int t = 0; t < 4; ++t) acc[t] = vzero;
    const _Float16* wrow = w16 + (size_t)(ct * 16 + ln) * C_DIM;
    #pragma unroll
    for (int kk = 0; kk < C_DIM; kk += 32) {
      v16h af = ld_frag(wrow + kk + aoff, wrow + kk + 16 + aoff);
      #pragma unroll
      for (int t = 0; t < 4; ++t) {
        v16h bf = ld_frag(hs + (t * 16 + ln) * QROW + kk + aoff,
                          hs + (t * 16 + ln) * QROW + kk + 16 + aoff);
        acc[t] = wmma_f16(af, bf, acc[t]);
      }
    }
    #pragma unroll
    for (int t = 0; t < 4; ++t) {
      const int i = i0 + t * 16 + ln;
      if (!transpose_store) {
        FragU pk;
        _Float16 tmp[8];
        #pragma unroll
        for (int r = 0; r < 8; ++r)
          tmp[r] = (_Float16)((acc[t][r] + bias[ct * 16 + r + 8 * hl]) * scale);
        pk.u[0] = *(const uint4*)tmp;
        *(uint4*)&outp[((size_t)b * N_TOK + i) * C_DIM + ct * 16 + 8 * hl] = pk.u[0];
      } else {
        #pragma unroll
        for (int r = 0; r < 8; ++r) {
          int co = ct * 16 + r + 8 * hl;
          outp[((size_t)b * C_DIM + co) * N_TOK + i] =
              (_Float16)((acc[t][r] + bias[co]) * scale);
        }
      }
    }
  }
}

// ---------------------------------------------------------------------------
// K4: flash attention, async double-buffered k/v tiles.
// Block = 64 query rows (4 waves x 16), keys streamed 32 at a time.
// q,k: [b][n][c] f16 ; v: [b][c][n] f16 ; out hbuf: [b][n][c] f16
// ---------------------------------------------------------------------------
#define BM 64
#define BN 32
__global__ __launch_bounds__(128) void flash_kernel(const _Float16* __restrict__ qg,
                                                    const _Float16* __restrict__ kg,
                                                    const _Float16* __restrict__ vg,
                                                    _Float16* __restrict__ hbuf) {
  extern __shared__ __align__(16) char smem[];
  _Float16* qs = (_Float16*)smem;                    // [64][QROW]
  _Float16* ks0 = qs + BM * QROW;                    // [32][QROW] x2
  _Float16* ks1 = ks0 + BN * QROW;
  _Float16* vs0 = ks1 + BN * QROW;                   // [512][VROW] x2
  _Float16* vs1 = vs0 + C_DIM * VROW;
  _Float16* pb = vs1 + C_DIM * VROW;                 // [4][16][VROW]

  const int tid = threadIdx.x;
  const int lane = tid & 31;
  const int wid = tid >> 5;
  const int hl = lane >> 4;
  const int ln = lane & 15;
  const int aoff = hl * 8;
  const int koff = hl * 16;
  const int b = blockIdx.x >> 6;
  const int istrip = blockIdx.x & 63;

  // stage q strip (regular loads, padded rows)
  for (int t = tid; t < BM * 64; t += 128) {
    int r = t >> 6, seg = (t & 63) * 8;
    *(uint4*)(qs + r * QROW + seg) =
        *(const uint4*)(qg + ((size_t)b * N_TOK + istrip * BM + r) * C_DIM + seg);
  }

  _Float16* const ksb[2] = {ks0, ks1};
  _Float16* const vsb[2] = {vs0, vs1};

  auto issue_tile = [&](int j0, int buf) {
    unsigned kbase = (unsigned)(uintptr_t)ksb[buf];
    unsigned vbase = (unsigned)(uintptr_t)vsb[buf];
    const _Float16* ksrc = kg + ((size_t)b * N_TOK + j0) * C_DIM;
    for (int t = tid; t < BN * 64; t += 128) {  // 2048 x 16B
      int r = t >> 6, seg = (t & 63) * 8;
      async_b128(kbase + (unsigned)(r * QROW + seg) * 2u, ksrc + r * C_DIM + seg);
    }
    for (int t = tid; t < C_DIM * 4; t += 128) {  // 2048 x 16B
      int c = t >> 2, ch = (t & 3) * 8;
      async_b128(vbase + (unsigned)(c * VROW + ch) * 2u,
                 vg + ((size_t)b * C_DIM + c) * N_TOK + j0 + ch);
    }
  };

  issue_tile(0, 0);

  const v8f vzero = {0.f, 0.f, 0.f, 0.f, 0.f, 0.f, 0.f, 0.f};
  v8f acc[32];
  #pragma unroll
  for (int ct = 0; ct < 32; ++ct) acc[ct] = vzero;
  float mrow[8], lrow[8], alpha[8];
  #pragma unroll
  for (int r = 0; r < 8; ++r) { mrow[r] = -1e30f; lrow[r] = 0.f; }

  _Float16* mypb = pb + wid * (16 * VROW);
  const _Float16* qrow = qs + (wid * 16 + ln) * QROW;

  int pp = 0;
  for (int j0 = 0; j0 < N_TOK; j0 += BN, pp ^= 1) {
    wait_async0();       // my tile-j0 copies done
    __syncthreads();     // everyone's copies done; prev buffer free
    if (j0 + BN < N_TOK) issue_tile(j0 + BN, pp ^ 1);
    const _Float16* ks = ksb[pp];
    const _Float16* vs = vsb[pp];

    // ---- S = q . k^T over K=512 (two 16x16 j-tiles per wave) ----
    v8f s0 = vzero, s1 = vzero;
    #pragma unroll
    for (int kk = 0; kk < C_DIM; kk += 32) {
      v16h af = ld_frag(qrow + kk + aoff, qrow + kk + 16 + aoff);
      v16h b0 = ld_frag(ks + ln * QROW + kk + aoff,
                        ks + ln * QROW + kk + 16 + aoff);
      v16h b1 = ld_frag(ks + (16 + ln) * QROW + kk + aoff,
                        ks + (16 + ln) * QROW + kk + 16 + aoff);
      s0 = wmma_f16(af, b0, s0);
      s1 = wmma_f16(af, b1, s1);
    }

    // ---- online softmax per row ----
    #pragma unroll
    for (int r = 0; r < 8; ++r) {
      float e0 = s0[r], e1 = s1[r];
      float mx = fmaxf(e0, e1);
      mx = fmaxf(mx, __shfl_xor(mx, 1, 32));
      mx = fmaxf(mx, __shfl_xor(mx, 2, 32));
      mx = fmaxf(mx, __shfl_xor(mx, 4, 32));
      mx = fmaxf(mx, __shfl_xor(mx, 8, 32));
      float mnew = fmaxf(mrow[r], mx);
      alpha[r] = __expf(mrow[r] - mnew);
      float p0 = __expf(e0 - mnew);
      float p1 = __expf(e1 - mnew);
      float rs = p0 + p1;
      rs += __shfl_xor(rs, 1, 32);
      rs += __shfl_xor(rs, 2, 32);
      rs += __shfl_xor(rs, 4, 32);
      rs += __shfl_xor(rs, 8, 32);
      lrow[r] = lrow[r] * alpha[r] + rs;
      mrow[r] = mnew;
      int M = r + 8 * hl;
      mypb[M * VROW + ln] = (_Float16)p0;
      mypb[M * VROW + 16 + ln] = (_Float16)p1;
    }
    #pragma unroll
    for (int ct = 0; ct < 32; ++ct) {
      #pragma unroll
      for (int r = 0; r < 8; ++r) acc[ct][r] *= alpha[r];
    }

    // ---- O += P . V^T across all 512 channels ----
    v16h pf = ld_frag(mypb + ln * VROW + aoff, mypb + ln * VROW + 16 + aoff);
    #pragma unroll
    for (int ct = 0; ct < 32; ++ct) {
      v16h bf = ld_frag(vs + (ct * 16 + ln) * VROW + aoff,
                        vs + (ct * 16 + ln) * VROW + 16 + aoff);
      acc[ct] = wmma_f16(pf, bf, acc[ct]);
    }
  }

  // ---- finalize ----
  float inv[8];
  #pragma unroll
  for (int r = 0; r < 8; ++r) inv[r] = 1.0f / lrow[r];
  const int ig = istrip * BM + wid * 16;
  #pragma unroll
  for (int ct = 0; ct < 32; ++ct) {
    #pragma unroll
    for (int r = 0; r < 8; ++r) {
      int M = r + 8 * hl;
      hbuf[((size_t)b * N_TOK + ig + M) * C_DIM + ct * 16 + ln] =
          (_Float16)(acc[ct][r] * inv[r]);
    }
  }
}

// ---------------------------------------------------------------------------
// K5: final projection + residual, fp32 out[b][c][n] = x + Wp.h + bp
// ---------------------------------------------------------------------------
__global__ __launch_bounds__(128) void proj_out_kernel(const _Float16* __restrict__ hb,
                                                       const _Float16* __restrict__ w16,
                                                       const float* __restrict__ bias,
                                                       const float* __restrict__ x,
                                                       float* __restrict__ out) {
  extern __shared__ __align__(16) char smem[];
  _Float16* hs = (_Float16*)smem;  // [64][QROW]
  const int tid = threadIdx.x;
  const int lane = tid & 31;
  const int wid = tid >> 5;
  const int hl = lane >> 4;
  const int ln = lane & 15;
  const int aoff = hl * 8;
  const int koff = hl * 16;
  const int b = blockIdx.x >> 6;
  const int i0 = (blockIdx.x & 63) << 6;

  for (int t = tid; t < 64 * 64; t += 128) {
    int r = t >> 6, seg = (t & 63) * 8;
    *(uint4*)(hs + r * QROW + seg) =
        *(const uint4*)(hb + ((size_t)b * N_TOK + i0 + r) * C_DIM + seg);
  }
  __syncthreads();

  for (int ct = wid; ct < 32; ct += 4) {
    const v8f vzero = {0.f, 0.f, 0.f, 0.f, 0.f, 0.f, 0.f, 0.f};
    v8f acc[4];
    #pragma unroll
    for (int t = 0; t < 4; ++t) acc[t] = vzero;
    const _Float16* wrow = w16 + (size_t)(ct * 16 + ln) * C_DIM;
    #pragma unroll
    for (int kk = 0; kk < C_DIM; kk += 32) {
      v16h af = ld_frag(wrow + kk + aoff, wrow + kk + 16 + aoff);
      #pragma unroll
      for (int t = 0; t < 4; ++t) {
        v16h bf = ld_frag(hs + (t * 16 + ln) * QROW + kk + aoff,
                          hs + (t * 16 + ln) * QROW + kk + 16 + aoff);
        acc[t] = wmma_f16(af, bf, acc[t]);
      }
    }
    #pragma unroll
    for (int t = 0; t < 4; ++t) {
      const int i = i0 + t * 16 + ln;
      #pragma unroll
      for (int r = 0; r < 8; ++r) {
        int co = ct * 16 + r + 8 * hl;
        size_t idx = ((size_t)b * C_DIM + co) * N_TOK + i;
        out[idx] = x[idx] + acc[t][r] + bias[co];
      }
    }
  }
}

// ---------------------------------------------------------------------------
// host launch
// ---------------------------------------------------------------------------
extern "C" void kernel_launch(void* const* d_in, const int* in_sizes, int n_in,
                              void* d_out, int out_size, void* d_ws, size_t ws_size,
                              hipStream_t stream) {
  (void)in_sizes; (void)n_in; (void)out_size; (void)ws_size;
  const float* x   = (const float*)d_in[0];
  const float* gnw = (const float*)d_in[1];
  const float* gnb = (const float*)d_in[2];
  const float* wq  = (const float*)d_in[3];
  const float* bq  = (const float*)d_in[4];
  const float* wk  = (const float*)d_in[5];
  const float* bk  = (const float*)d_in[6];
  const float* wv  = (const float*)d_in[7];
  const float* bv  = (const float*)d_in[8];
  const float* wp  = (const float*)d_in[9];
  const float* bp  = (const float*)d_in[10];
  float* out = (float*)d_out;
  char* ws = (char*)d_ws;

  const size_t WSZ = (size_t)C_DIM * C_DIM * sizeof(_Float16);          // 512KB
  const size_t BUF = (size_t)B_DIM * N_TOK * C_DIM * sizeof(_Float16);  // 16MB
  float*    stats = (float*)ws;
  _Float16* wq16  = (_Float16*)(ws + 1024);
  _Float16* wk16  = (_Float16*)(ws + 1024 + WSZ);
  _Float16* wv16  = (_Float16*)(ws + 1024 + 2 * WSZ);
  _Float16* wp16  = (_Float16*)(ws + 1024 + 3 * WSZ);
  _Float16* hn    = (_Float16*)(ws + 1024 + 4 * WSZ);  // reused as attn output
  _Float16* qb    = (_Float16*)(ws + 1024 + 4 * WSZ + BUF);
  _Float16* kb    = (_Float16*)(ws + 1024 + 4 * WSZ + 2 * BUF);
  _Float16* vb    = (_Float16*)(ws + 1024 + 4 * WSZ + 3 * BUF);

  const int wn = C_DIM * C_DIM;
  wconv_kernel<<<wn / 256, 256, 0, stream>>>(wq, wq16, wn);
  wconv_kernel<<<wn / 256, 256, 0, stream>>>(wk, wk16, wn);
  wconv_kernel<<<wn / 256, 256, 0, stream>>>(wv, wv16, wn);
  wconv_kernel<<<wn / 256, 256, 0, stream>>>(wp, wp16, wn);

  gn_stats_kernel<<<B_DIM * NGRP, 256, 0, stream>>>(x, stats);
  hn_kernel<<<B_DIM * 16 * 128, dim3(32, 8), 0, stream>>>(x, stats, gnw, gnb, hn);

  const size_t proj_smem = (size_t)64 * QROW * sizeof(_Float16);  // 66.5KB
  const int proj_grid = B_DIM * (N_TOK / 64);
  proj_kernel<<<proj_grid, 128, proj_smem, stream>>>(hn, wq16, bq, qb, Q_SCALE, 0);
  proj_kernel<<<proj_grid, 128, proj_smem, stream>>>(hn, wk16, bk, kb, 1.0f, 0);
  proj_kernel<<<proj_grid, 128, proj_smem, stream>>>(hn, wv16, bv, vb, 1.0f, 1);

  const size_t flash_smem =
      (size_t)(BM * QROW + 2 * BN * QROW + 2 * C_DIM * VROW + 4 * 16 * VROW) *
      sizeof(_Float16);  // ~220KB
  flash_kernel<<<B_DIM * (N_TOK / BM), 128, flash_smem, stream>>>(qb, kb, vb, hn);

  proj_out_kernel<<<proj_grid, 128, proj_smem, stream>>>(hn, wp16, bp, x, out);
}
